// GANLayer_52097953300845
// MI455X (gfx1250) — matrix operand order repaired and checked by reading, not compiled, not run on hardware
//
#include <hip/hip_runtime.h>
#include <hip/hip_bf16.h>
#include <utility>

#define NND 1147
#define HD  512
#define NHD 8

typedef __bf16 bf16;
typedef bf16  v16bf __attribute__((ext_vector_type(16)));
typedef float v8f   __attribute__((ext_vector_type(8)));

__device__ __forceinline__ bf16 f2bf(float f) {
  unsigned u = __builtin_bit_cast(unsigned, f);
  u = (u + 0x7fffu + ((u >> 16) & 1u)) >> 16;
  unsigned short s = (unsigned short)u;
  return __builtin_bit_cast(bf16, s);
}
__device__ __forceinline__ unsigned pack2bf(float a, float b) {
  unsigned ua = (unsigned)__builtin_bit_cast(unsigned short, f2bf(a));
  unsigned ub = (unsigned)__builtin_bit_cast(unsigned short, f2bf(b));
  return ua | (ub << 16);
}

// ---------------- tiled bf16 WMMA GEMM: C = act(op(A)@B + bias) (+resid) ----
// A: [M,K] row-major (transA=0) or [K,M] row-major (transA=1, i.e. C = A^T B)
// B: [K,Nc] row-major, C/resid: [M,Nc] row-major, bias: [Nc]
//
// LDS tiles live in WMMA fragment order with double buffering; staging is
// strength-reduced to base-pointer + constant strides so global addresses are
// incremented, not recomputed, and LDS stores get immediate offsets.
//
// A fragment map (16x32, 16-bit): lane = (m&15) + ((k&15)>=8 ? 16:0),
//                                 j = (k&7) + (k>=16 ? 8:0)
// B fragment map (32x16, 16-bit): lane = (n&15) + ((k&15)>=8 ? 16:0),
//                                 j = (k&7) + (k>=16 ? 8:0)
__global__ void __launch_bounds__(128) k_gemm(
    const float* __restrict__ A, const float* __restrict__ B,
    const float* __restrict__ bias, const float* __restrict__ resid,
    float* __restrict__ C, int M, int Nc, int K, int transA, int relu)
{
  __shared__ __align__(32) bf16 As2[2][4][32][16];
  __shared__ __align__(32) bf16 Bs2[2][4][32][16];
  const int tid  = threadIdx.x;
  const int bm   = blockIdx.y * 64, bn = blockIdx.x * 64;
  const int wave = tid >> 5, lane = tid & 31;
  const int rs0  = (wave >> 1) * 2;   // A row-subtile base for this wave
  const int cs0  = (wave & 1) * 2;    // B col-subtile base for this wave

  const bool fullM = (bm + 64 <= M);
  const bool fullN = (bn + 64 <= Nc);
  const bool evenK = (K & 1) == 0;

  // ---- per-thread staging constants (loop-invariant) ----
  const int m0  = tid >> 4;            // A: row group 0..7
  const int kkA = (tid & 15) * 2;      // A: K-pair base 0..30 (const!)
  const int hiA = ((kkA & 15) >= 8) ? 16 : 0;
  const int jA  = (kkA & 7) + ((kkA >= 16) ? 8 : 0);
  const int nnB = tid & 63;            // B: column (const!)
  const int k1B = (tid >> 6) * 2;      // B: K-pair sub-base 0 or 2
  const int mT  = tid & 63;            // A^T: column (const!)
  const int t1T = tid >> 6;

  auto stageA = [&](int k0, int buf) {
    if (!transA) {
      char* base = (char*)&As2[buf][0][m0 + hiA][jA];
      const float* p = A + (size_t)(bm + m0) * K + (k0 + kkA);
      if (fullM && (k0 + 32 <= K)) {
        if (evenK) {
#pragma unroll
          for (int i = 0; i < 8; ++i) {
            float2 v = *(const float2*)p;  p += 8 * (size_t)K;
            *(unsigned*)(base + (i >> 1) * 1024 + (i & 1) * 256) = pack2bf(v.x, v.y);
          }
        } else {
#pragma unroll
          for (int i = 0; i < 8; ++i) {
            float va = p[0], vb = p[1];  p += 8 * (size_t)K;
            *(unsigned*)(base + (i >> 1) * 1024 + (i & 1) * 256) = pack2bf(va, vb);
          }
        }
      } else {
        const bool ka = (k0 + kkA) < K, kb = (k0 + kkA + 1) < K;
#pragma unroll
        for (int i = 0; i < 8; ++i) {
          bool mok = (bm + m0 + i * 8) < M;
          float va = (mok && ka) ? p[0] : 0.f;
          float vb = (mok && kb) ? p[1] : 0.f;
          p += 8 * (size_t)K;
          *(unsigned*)(base + (i >> 1) * 1024 + (i & 1) * 256) = pack2bf(va, vb);
        }
      }
    } else {      // C = A^T B: A is [K,M]; per-thread column mT is constant
      char* tbase = (char*)&As2[buf][mT >> 4][0][0];
      const bool mok = (bm + mT) < M;
      const float* p = A + (bm + mT) + (size_t)(k0 + 2 * t1T) * M;
#pragma unroll
      for (int i = 0; i < 8; ++i) {
        int kk = i * 4 + 2 * t1T;
        bool ka = (k0 + kk) < K, kb = (k0 + kk + 1) < K;
        float va = (mok && ka) ? p[0] : 0.f;
        float vb = (mok && kb) ? p[(size_t)M] : 0.f;
        p += 4 * (size_t)M;
        int lt = (mT & 15) + (((kk & 15) >= 8) ? 16 : 0);
        int j  = (kk & 7) + ((kk >= 16) ? 8 : 0);
        *(unsigned*)(tbase + (lt * 16 + j) * 2) = pack2bf(va, vb);
      }
    }
  };

  auto stageB = [&](int k0, int buf) {
    char* bbase = (char*)&Bs2[buf][nnB >> 4][nnB & 15][0];
    const float* p = B + (bn + nnB) + (size_t)(k0 + k1B) * Nc;
    if (fullN && (k0 + 32 <= K)) {
#pragma unroll
      for (int i = 0; i < 8; ++i) {
        int kk = i * 4 + k1B;
        float va = p[0], vb = p[(size_t)Nc];
        p += 4 * (size_t)Nc;
        *(unsigned*)(bbase + (((kk & 15) >= 8) ? 512 : 0) + ((kk & 7) + ((kk >= 16) ? 8 : 0)) * 2) = pack2bf(va, vb);
      }
    } else {
      const bool nok = (bn + nnB) < Nc;
#pragma unroll
      for (int i = 0; i < 8; ++i) {
        int kk = i * 4 + k1B;
        bool ka = (k0 + kk) < K, kb = (k0 + kk + 1) < K;
        float va = (nok && ka) ? p[0] : 0.f;
        float vb = (nok && kb) ? p[(size_t)Nc] : 0.f;
        p += 4 * (size_t)Nc;
        *(unsigned*)(bbase + (((kk & 15) >= 8) ? 512 : 0) + ((kk & 7) + ((kk >= 16) ? 8 : 0)) * 2) = pack2bf(va, vb);
      }
    }
  };

  v8f acc00, acc01, acc10, acc11;
#pragma unroll
  for (int i = 0; i < 8; ++i) { acc00[i]=0.f; acc01[i]=0.f; acc10[i]=0.f; acc11[i]=0.f; }

  const int nChunks = (K + 31) >> 5;
  stageA(0, 0); stageB(0, 0);
  __syncthreads();

  for (int c = 0; c < nChunks; ++c) {
    const int buf = c & 1;
    // fragment loads: contiguous 32B per lane -> 2x ds_load_b128 each
    v16bf a0 = *reinterpret_cast<const v16bf*>(&As2[buf][rs0    ][lane][0]);
    v16bf a1 = *reinterpret_cast<const v16bf*>(&As2[buf][rs0 + 1][lane][0]);
    v16bf b0 = *reinterpret_cast<const v16bf*>(&Bs2[buf][cs0    ][lane][0]);
    v16bf b1 = *reinterpret_cast<const v16bf*>(&Bs2[buf][cs0 + 1][lane][0]);

    // stage next chunk into the other buffer while WMMAs run
    if (c + 1 < nChunks) {
      stageA((c + 1) * 32, buf ^ 1);
      stageB((c + 1) * 32, buf ^ 1);
      if (c + 2 < nChunks) {       // prefetch chunk c+2 toward L2
        int kp = (c + 2) * 32;
        if (!transA)
          __builtin_prefetch(&A[(size_t)(bm + (tid & 63)) * K + kp], 0, 3);
        else
          __builtin_prefetch(&A[(size_t)(kp + (tid >> 6)) * M + bm + (tid & 63)], 0, 3);
        __builtin_prefetch(&B[(size_t)(kp + (tid >> 6)) * Nc + bn + (tid & 63)], 0, 3);
      }
    }

    acc00 = __builtin_amdgcn_wmma_f32_16x16x32_bf16(false, a0, false, b0, (short)0, acc00, false, false);
    acc01 = __builtin_amdgcn_wmma_f32_16x16x32_bf16(false, a0, false, b1, (short)0, acc01, false, false);
    acc10 = __builtin_amdgcn_wmma_f32_16x16x32_bf16(false, a1, false, b0, (short)0, acc10, false, false);
    acc11 = __builtin_amdgcn_wmma_f32_16x16x32_bf16(false, a1, false, b1, (short)0, acc11, false, false);
    __syncthreads();
  }

  // ---- store D (C/D layout: VGPR r <-> rows r / r+8 of 16x16 tile)
  const int mh = (lane < 16) ? 0 : 8;
  const int nl = lane & 15;
#pragma unroll
  for (int r = 0; r < 2; ++r) {
    const int gmBase = bm + (rs0 + r) * 16 + mh;
    const size_t rowOff = (size_t)gmBase * Nc;
#pragma unroll
    for (int s = 0; s < 2; ++s) {
      const int gn = bn + (cs0 + s) * 16 + nl;
      if (gn >= Nc) continue;
      const float bv = bias ? bias[gn] : 0.f;
      v8f acc = (r == 0) ? ((s == 0) ? acc00 : acc01) : ((s == 0) ? acc10 : acc11);
      float* cp = C + rowOff + gn;
      const float* rp = resid ? resid + rowOff + gn : nullptr;
#pragma unroll
      for (int vg = 0; vg < 8; ++vg) {
        if (gmBase + vg < M) {
          float v = acc[vg] + bv;
          if (relu)  v = fmaxf(v, 0.f);
          if (resid) v += rp[(size_t)vg * Nc];
          cp[(size_t)vg * Nc] = v;
        }
      }
    }
  }
}

// ---------------- LayerNorm over rows of width 512 (one wave32 per row) -----
__global__ void __launch_bounds__(256) k_ln(const float* __restrict__ X,
                                            float* __restrict__ Y, int rows)
{
  int lane = threadIdx.x & 31, wave = threadIdx.x >> 5;
  int row = blockIdx.x * 8 + wave;
  if (row >= rows) return;
  const float* x = X + (size_t)row * HD;
  float v[16];
  float s = 0.f;
#pragma unroll
  for (int j = 0; j < 16; ++j) { v[j] = x[lane + 32 * j]; s += v[j]; }
#pragma unroll
  for (int m = 1; m < 32; m <<= 1) s += __shfl_xor(s, m, 32);
  float mean = s * (1.0f / HD);
  float q = 0.f;
#pragma unroll
  for (int j = 0; j < 16; ++j) { float d = v[j] - mean; q += d * d; }
#pragma unroll
  for (int m = 1; m < 32; m <<= 1) q += __shfl_xor(q, m, 32);
  float inv = rsqrtf(q * (1.0f / HD) + 1e-5f);
  float* y = Y + (size_t)row * HD;
#pragma unroll
  for (int j = 0; j < 16; ++j) y[lane + 32 * j] = (v[j] - mean) * inv;
}

// ---------------- fills ----------------
__global__ void k_fillf(float* p, float v, size_t n) {
  size_t i = (size_t)blockIdx.x * blockDim.x + threadIdx.x;
  if (i < n) p[i] = v;
}
__global__ void k_fillu(unsigned* p, unsigned v, size_t n) {
  size_t i = (size_t)blockIdx.x * blockDim.x + threadIdx.x;
  if (i < n) p[i] = v;
}

// ---------------- rank-1 edge feature projection: u=W_e0@We1, c=b_e@We1 ----
__global__ void k_edge_uc(const float* __restrict__ We0, const float* __restrict__ be,
                          const float* __restrict__ We1, float* u, float* c)
{
  int j = blockIdx.x * blockDim.x + threadIdx.x;
  if (j >= HD) return;
  float su = 0.f, sc = 0.f;
  for (int i = 0; i < HD; ++i) {
    float w = We1[(size_t)i * HD + j];
    su += We0[i] * w; sc += be[i] * w;
  }
  u[j] = su; c[j] = sc;
}

// ---------------- attention logits (one wave32 per edge) -------------------
__global__ void __launch_bounds__(256) k_edge_logits(
    const float* __restrict__ q, const float* __restrict__ k,
    const float* __restrict__ u, const float* __restrict__ c,
    const float* __restrict__ ew, const int* __restrict__ src,
    const int* __restrict__ dst, float* __restrict__ logits, int P, int useEp)
{
  int lane = threadIdx.x & 31, wave = threadIdx.x >> 5;
  int p = blockIdx.x * 8 + wave;
  if (p >= P) return;
  int s = src[p], d = dst[p];
  const float* kr = k + (size_t)s * HD;
  const float* qr = q + (size_t)d * HD;
  int base = lane * 16;            // lane's 16 elems live in head lane>>2
  float s1 = 0.f, s2 = 0.f;
#pragma unroll
  for (int j = 0; j < 16; ++j) {
    int el = base + j;
    float pr = kr[el] * qr[el];
    if (useEp) { s1 += pr * u[el]; s2 += pr * c[el]; }
    else s1 += pr;
  }
  s1 += __shfl_xor(s1, 1, 32); s1 += __shfl_xor(s1, 2, 32);
  s2 += __shfl_xor(s2, 1, 32); s2 += __shfl_xor(s2, 2, 32);
  if ((lane & 3) == 0) {
    int h = lane >> 2;
    float lg = useEp ? 0.125f * (ew[p] * s1 + s2) : 0.125f * s1;
    logits[(size_t)p * NHD + h] = lg;
  }
}

// ---------------- segment softmax (atomics) --------------------------------
__device__ __forceinline__ unsigned f2key(float f) {
  unsigned b = __builtin_bit_cast(unsigned, f);
  return (b & 0x80000000u) ? ~b : (b | 0x80000000u);
}
__device__ __forceinline__ float key2f(unsigned k) {
  unsigned b = (k & 0x80000000u) ? (k & 0x7fffffffu) : ~k;
  return __builtin_bit_cast(float, b);
}
__global__ void k_segmax(const float* __restrict__ logits, const int* __restrict__ dst,
                         unsigned* __restrict__ mk, int P)
{
  int idx = blockIdx.x * blockDim.x + threadIdx.x;
  if (idx >= P * NHD) return;
  int p = idx >> 3, h = idx & 7;
  atomicMax(&mk[(size_t)dst[p] * NHD + h], f2key(logits[idx]));
}
__global__ void k_m2v(const unsigned* __restrict__ mk, float* __restrict__ mv, int n8) {
  int i = blockIdx.x * blockDim.x + threadIdx.x;
  if (i >= n8) return;
  unsigned k = mk[i];
  mv[i] = (k == 0u) ? 0.f : key2f(k);   // nodes with no in-edges -> 0
}
__global__ void k_expden(float* __restrict__ logits, const int* __restrict__ dst,
                         const float* __restrict__ mv, float* __restrict__ den, int P)
{
  int idx = blockIdx.x * blockDim.x + threadIdx.x;
  if (idx >= P * NHD) return;
  int p = idx >> 3, h = idx & 7;
  float a = __expf(logits[idx] - mv[(size_t)dst[p] * NHD + h]);
  logits[idx] = a;
  atomicAdd(&den[(size_t)dst[p] * NHD + h], a);
}
__global__ void __launch_bounds__(256) k_agg(
    const float* __restrict__ a, const float* __restrict__ den,
    const float* __restrict__ v, const int* __restrict__ src,
    const int* __restrict__ dst, float* __restrict__ agg, int P)
{
  int lane = threadIdx.x & 31, wave = threadIdx.x >> 5;
  int p = blockIdx.x * 8 + wave;
  if (p >= P) return;
  int s = src[p], d = dst[p];
  int h = lane >> 2;
  float w = a[(size_t)p * NHD + h] / (den[(size_t)d * NHD + h] + 1e-9f);
  int base = lane * 16;
#pragma unroll
  for (int j = 0; j < 16; ++j) {
    int el = base + j;
    atomicAdd(&agg[(size_t)d * HD + el], w * v[(size_t)s * HD + el]);
  }
}

// ---------------- Laplacian-PE support -------------------------------------
__global__ void k_adj(const int* __restrict__ src, const int* __restrict__ dst,
                      float* __restrict__ A, int n, int P)
{
  int p = blockIdx.x * blockDim.x + threadIdx.x;
  if (p >= P) return;
  A[(size_t)src[p] * n + dst[p]] = 1.0f;
  A[(size_t)dst[p] * n + src[p]] = 1.0f;
}
__global__ void __launch_bounds__(256) k_deg(const float* __restrict__ A,
                                             float* __restrict__ dinv,
                                             float* __restrict__ v0, int n)
{
  int lane = threadIdx.x & 31, wave = threadIdx.x >> 5;
  int row = blockIdx.x * 8 + wave;
  if (row >= n) return;
  float s = 0.f;
  for (int j = lane; j < n; j += 32) s += A[(size_t)row * n + j];
#pragma unroll
  for (int m = 1; m < 32; m <<= 1) s += __shfl_xor(s, m, 32);
  if (lane == 0) { dinv[row] = rsqrtf(fmaxf(s, 1.0f)); v0[row] = sqrtf(s); }
}
__global__ void k_nacc(const float* v0, float* nrm, int n) {
  int i = blockIdx.x * blockDim.x + threadIdx.x;
  if (i < n) atomicAdd(nrm, v0[i] * v0[i]);
}
__global__ void k_nscale(float* v0, const float* nrm, int n) {
  int i = blockIdx.x * blockDim.x + threadIdx.x;
  if (i < n) v0[i] *= rsqrtf(nrm[0] + 1e-20f);
}
// Mm = 2I - L = I + dinv_i * A * dinv_j
__global__ void k_bM(const float* __restrict__ A, const float* __restrict__ dinv,
                     float* __restrict__ Mm, int n, size_t total)
{
  size_t idx = (size_t)blockIdx.x * blockDim.x + threadIdx.x;
  if (idx >= total) return;
  int i = (int)(idx / n), j = (int)(idx % n);
  Mm[idx] = ((i == j) ? 1.0f : 0.0f) + dinv[i] * A[idx] * dinv[j];
}
__device__ __forceinline__ float hrand(unsigned x) {
  x ^= x >> 16; x *= 0x7feb352du; x ^= x >> 15; x *= 0x846ca68bu; x ^= x >> 16;
  return (float)(x & 0xffffffu) * (1.0f / 16777216.0f) - 0.5f;
}
__global__ void k_initQ(float* Q, size_t total) {
  size_t i = (size_t)blockIdx.x * blockDim.x + threadIdx.x;
  if (i < total) Q[i] = hrand((unsigned)i * 2654435761u + 1234567u);
}
// T = 1.5*I - 0.5*S (Newton-Schulz combine)
__global__ void k_nsT(const float* __restrict__ S, float* __restrict__ T) {
  int idx = blockIdx.x * blockDim.x + threadIdx.x;
  if (idx >= HD * HD) return;
  int i = idx >> 9, j = idx & 511;
  T[idx] = ((i == j) ? 1.5f : 0.0f) - 0.5f * S[idx];
}
__global__ void __launch_bounds__(256) k_coldot(const float* __restrict__ v0,
                                                const float* __restrict__ Q,
                                                float* __restrict__ coef, int n)
{
  int lane = threadIdx.x & 31, wave = threadIdx.x >> 5;
  int col = blockIdx.x * 8 + wave;
  if (col >= HD) return;
  float s = 0.f;
  for (int i = lane; i < n; i += 32) s += v0[i] * Q[(size_t)i * HD + col];
#pragma unroll
  for (int m = 1; m < 32; m <<= 1) s += __shfl_xor(s, m, 32);
  if (lane == 0) coef[col] = s;
}
__global__ void k_coldef(float* __restrict__ Q, const float* __restrict__ v0,
                         const float* __restrict__ coef, size_t total)
{
  size_t idx = (size_t)blockIdx.x * blockDim.x + threadIdx.x;
  if (idx >= total) return;
  int i = (int)(idx >> 9), j = (int)(idx & 511);
  Q[idx] -= coef[j] * v0[i];
}

// ---------------- host orchestration ---------------------------------------
extern "C" void kernel_launch(void* const* d_in, const int* in_sizes, int n_in,
                              void* d_out, int out_size, void* d_ws, size_t ws_size,
                              hipStream_t stream)
{
  const int n = NND, D = HD;
  const int P = in_sizes[2];

  const float* H     = (const float*)d_in[0];
  const float* ew    = (const float*)d_in[1];
  const int*   esrc  = (const int*)d_in[2];
  const int*   edst  = (const int*)d_in[3];
  const float* W_lin = (const float*)d_in[4];
  const float* b_lin = (const float*)d_in[5];
  const float* W_e   = (const float*)d_in[6];
  const float* b_e   = (const float*)d_in[7];
  const float* W_h   = (const float*)d_in[8];
  const float* b_h   = (const float*)d_in[9];
  const float* W_lap = (const float*)d_in[10];
  const float* b_lap = (const float*)d_in[11];
  const float* Wq1 = (const float*)d_in[12];
  const float* Wk1 = (const float*)d_in[13];
  const float* Wv1 = (const float*)d_in[14];
  const float* We1 = (const float*)d_in[15];
  const float* Wo1 = (const float*)d_in[16];
  // d_in[17] Woe1, d_in[20] We1a, d_in[21] We1b: dead — edge output path discarded
  const float* Wf1a = (const float*)d_in[18];
  const float* Wf1b = (const float*)d_in[19];
  const float* Wq2 = (const float*)d_in[22];
  const float* Wk2 = (const float*)d_in[23];
  const float* Wv2 = (const float*)d_in[24];
  const float* Wo2 = (const float*)d_in[25];
  const float* Wf2a = (const float*)d_in[26];
  const float* Wf2b = (const float*)d_in[27];
  (void)n_in; (void)ws_size;

  float* ws = (float*)d_ws;
  size_t off = 0;
  auto alloc = [&](size_t cnt) { float* p = ws + off; off += (cnt + 255) & ~(size_t)255; return p; };

  float* hl   = alloc((size_t)n * D);
  float* h0   = alloc((size_t)n * D);
  float* h2   = alloc((size_t)n * D);
  float* qb   = alloc((size_t)n * D);
  float* kb   = alloc((size_t)n * D);
  float* vb   = alloc((size_t)n * D);
  float* agg  = alloc((size_t)n * D);
  float* tb   = alloc((size_t)n * D);
  float* h1   = alloc((size_t)n * D);
  float* zb   = alloc((size_t)n * D);
  float* ffn  = alloc((size_t)n * 2 * D);
  float* Ab   = alloc((size_t)n * n);
  float* Mm   = alloc((size_t)n * n);
  float* Qa   = alloc((size_t)n * D);
  float* Qb2  = alloc((size_t)n * D);
  float* Sm   = alloc((size_t)D * D);
  float* Tm   = alloc((size_t)D * D);
  float* dinv = alloc(n);
  float* v0   = alloc(n);
  float* nrm  = alloc(1);
  float* uvec = alloc(D);
  float* cvec = alloc(D);
  float* coef = alloc(D);
  float* logits = alloc((size_t)P * NHD);
  float* mkey = alloc((size_t)n * NHD);
  float* mval = alloc((size_t)n * NHD);
  float* den  = alloc((size_t)n * NHD);

  auto gemm = [&](const float* A, const float* B, const float* bias,
                  const float* resid, float* C, int M, int Nc, int K,
                  int transA, int relu) {
    dim3 g((Nc + 63) / 64, (M + 63) / 64);
    k_gemm<<<g, 128, 0, stream>>>(A, B, bias, resid, C, M, Nc, K, transA, relu);
  };
  auto fillf = [&](float* p, float v, size_t cnt) {
    k_fillf<<<(unsigned)((cnt + 255) / 256), 256, 0, stream>>>(p, v, cnt);
  };
  auto ln = [&](const float* X, float* Y, int rows) {
    k_ln<<<(rows + 7) / 8, 256, 0, stream>>>(X, Y, rows);
  };
  auto deflate = [&](float* Q) {
    k_coldot<<<(D + 7) / 8, 256, 0, stream>>>(v0, Q, coef, n);
    k_coldef<<<(unsigned)(((size_t)n * D + 255) / 256), 256, 0, stream>>>(Q, v0, coef, (size_t)n * D);
  };
  float* Qc = Qa; float* Qo = Qb2;
  auto ns = [&]() {   // Qc <- Qc (1.5I - 0.5 Qc^T Qc)
    gemm(Qc, Qc, nullptr, nullptr, Sm, D, D, n, 1, 0);
    k_nsT<<<(D * D + 255) / 256, 256, 0, stream>>>(Sm, Tm);
    gemm(Qc, Tm, nullptr, nullptr, Qo, n, D, D, 0, 0);
    std::swap(Qc, Qo);
  };
  auto attn = [&](const float* qp, const float* kp, const float* vp, int useEp) {
    k_fillu<<<(unsigned)(((size_t)n * NHD + 255) / 256), 256, 0, stream>>>((unsigned*)mkey, 0u, (size_t)n * NHD);
    fillf(den, 0.f, (size_t)n * NHD);
    fillf(agg, 0.f, (size_t)n * D);
    k_edge_logits<<<(P + 7) / 8, 256, 0, stream>>>(qp, kp, uvec, cvec, ew, esrc, edst, logits, P, useEp);
    k_segmax<<<(P * NHD + 255) / 256, 256, 0, stream>>>(logits, edst, (unsigned*)mkey, P);
    k_m2v<<<(n * NHD + 255) / 256, 256, 0, stream>>>((unsigned*)mkey, mval, n * NHD);
    k_expden<<<(P * NHD + 255) / 256, 256, 0, stream>>>(logits, edst, mval, den, P);
    k_agg<<<(P + 7) / 8, 256, 0, stream>>>(logits, den, vp, esrc, edst, agg, P);
  };

  // ---- node embedding: h = (H @ W_lin + b_lin) @ W_h + b_h
  gemm(H, W_lin, b_lin, nullptr, hl, n, D, n, 0, 0);   // K = 1147
  gemm(hl, W_h, b_h, nullptr, h0, n, D, D, 0, 0);

  // ---- Laplacian PE via shifted subspace iteration on Mm = 2I - L
  fillf(Ab, 0.f, (size_t)n * n);
  k_adj<<<(P + 255) / 256, 256, 0, stream>>>(esrc, edst, Ab, n, P);
  k_deg<<<(n + 7) / 8, 256, 0, stream>>>(Ab, dinv, v0, n);
  fillf(nrm, 0.f, 1);
  k_nacc<<<(n + 255) / 256, 256, 0, stream>>>(v0, nrm, n);
  k_nscale<<<(n + 255) / 256, 256, 0, stream>>>(v0, nrm, n);
  k_bM<<<(unsigned)(((size_t)n * n + 255) / 256), 256, 0, stream>>>(Ab, dinv, Mm, n, (size_t)n * n);
  k_initQ<<<(unsigned)(((size_t)n * D + 255) / 256), 256, 0, stream>>>(Qc, (size_t)n * D);
  deflate(Qc); ns(); ns();
  for (int it = 0; it < 8; ++it) {
    gemm(Mm, Qc, nullptr, nullptr, Qo, n, D, n, 0, 0);  // Z = Mm @ Q
    std::swap(Qc, Qo);
    deflate(Qc);
    ns(); ns();
  }
  // h2 = h + pe @ W_lap + b_lap
  gemm(Qc, W_lap, b_lap, h0, h2, n, D, D, 0, 0);

  // ---- layer 1 (edge-conditioned graph transformer) ----
  gemm(h2, Wq1, nullptr, nullptr, qb, n, D, D, 0, 0);
  gemm(h2, Wk1, nullptr, nullptr, kb, n, D, D, 0, 0);
  gemm(h2, Wv1, nullptr, nullptr, vb, n, D, D, 0, 0);
  k_edge_uc<<<2, 256, 0, stream>>>(W_e, b_e, We1, uvec, cvec);
  attn(qb, kb, vb, 1);
  gemm(agg, Wo1, nullptr, h2, tb, n, D, D, 0, 0);
  ln(tb, h1, n);
  gemm(h1, Wf1a, nullptr, nullptr, ffn, n, 2 * D, D, 0, 1);      // relu
  gemm(ffn, Wf1b, nullptr, h1, tb, n, D, 2 * D, 0, 0);
  ln(tb, zb, n);

  // ---- layer 2 (plain graph transformer) ----
  gemm(zb, Wq2, nullptr, nullptr, qb, n, D, D, 0, 0);
  gemm(zb, Wk2, nullptr, nullptr, kb, n, D, D, 0, 0);
  gemm(zb, Wv2, nullptr, nullptr, vb, n, D, D, 0, 0);
  attn(qb, kb, vb, 0);
  gemm(agg, Wo2, nullptr, zb, tb, n, D, D, 0, 0);
  ln(tb, h1, n);
  gemm(h1, Wf2a, nullptr, nullptr, ffn, n, 2 * D, D, 0, 1);      // relu
  gemm(ffn, Wf2b, nullptr, h1, tb, n, D, 2 * D, 0, 0);
  ln(tb, (float*)d_out, n);
  (void)out_size;
}
